// _CrossAttentionLayer_48455821033916
// MI455X (gfx1250) — matrix operand order repaired and checked by reading, not compiled, not run on hardware
//
#include <hip/hip_runtime.h>
#include <hip/hip_bf16.h>
#include <stdint.h>

// ---------------------------------------------------------------------------
// Cross-attention layer for MI455X (gfx1250), wave32 + WMMA bf16 +
// async global->LDS staging (ASYNCcnt) for the K operand.
//   out, attn_mean = LN(q + (softmax(s*QK^T + bias) V) Wo^T + bo), mean_h(attn)
// Shapes: HIDDEN=512, HEADS=8, D_HEAD=64, N_L=2048, N_R=4096.
// ---------------------------------------------------------------------------

#define HIDDEN 512
#define HEADS 8
#define D_HEAD 64
#define NL 2048
#define NR 4096
#define ATT_SCALE 0.125f   // 1/sqrt(64)
#define LN_EPS 1e-5f

typedef __bf16 bf16_t;
typedef __attribute__((ext_vector_type(16))) __bf16 v16bf;
typedef __attribute__((ext_vector_type(8)))  __bf16 v8bf;
typedef __attribute__((ext_vector_type(8)))  float  v8f;

// ---------------- fragment loaders (layouts per CDNA5 ISA) -------
// A 16x32 bf16: lane -> row = lane&15; half j<8 : K = 8*hi + j
//                                   half j>=8: K = 16 + 8*hi + (j-8), hi=lane>>4
__device__ __forceinline__ v16bf load_a_frag_bf16(const bf16_t* base, int stride) {
  const int lane = threadIdx.x & 31;
  const bf16_t* p = base + (size_t)(lane & 15) * stride + 8 * (lane >> 4);
  v8bf lo = *(const v8bf*)(p);
  v8bf hi = *(const v8bf*)(p + 16);
  return __builtin_shufflevector(lo, hi, 0,1,2,3,4,5,6,7,8,9,10,11,12,13,14,15);
}

// B 32x16 bf16 (KxN): lane -> col n = lane&15; j<8: K = 8*hi + j, j>=8: K = 16 + 8*hi + (j-8)
// base points at element (K=0,N=0); element (k,n) = base[n*stride + k].
__device__ __forceinline__ v16bf load_b_frag_bf16(const bf16_t* base, int stride) {
  const int lane = threadIdx.x & 31;
  const bf16_t* p = base + (size_t)(lane & 15) * stride + 8 * (lane >> 4);
  v8bf lo = *(const v8bf*)(p);
  v8bf hi = *(const v8bf*)(p + 16);
  return __builtin_shufflevector(lo, hi, 0,1,2,3,4,5,6,7,8,9,10,11,12,13,14,15);
}

// Same fragments gathered from f32 memory with on-the-fly bf16 convert.
__device__ __forceinline__ v16bf load_a_frag_f32(const float* base, int stride) {
  const int lane = threadIdx.x & 31;
  const float* p = base + (size_t)(lane & 15) * stride + 8 * (lane >> 4);
  v16bf r;
#pragma unroll
  for (int j = 0; j < 8; ++j) r[j] = (bf16_t)p[j];
#pragma unroll
  for (int j = 0; j < 8; ++j) r[8 + j] = (bf16_t)p[16 + j];
  return r;
}

__device__ __forceinline__ v16bf load_b_frag_f32(const float* base, int stride) {
  const int lane = threadIdx.x & 31;
  const float* p = base + (size_t)(lane & 15) * stride + 8 * (lane >> 4);
  v16bf r;
#pragma unroll
  for (int j = 0; j < 16; ++j) r[j] = (bf16_t)p[j < 8 ? j : j + 8];
  return r;
}

#define WMMA_BF16(A, B, C) \
  __builtin_amdgcn_wmma_f32_16x16x32_bf16(false, (A), false, (B), (short)0, (C), false, false)

// -------- async global->LDS staging (gfx1250 VGLOBAL async ops, ASYNCcnt) ---
// Cooperative block copy of a contiguous BYTES region (BYTES % 4096 == 0;
// 256 threads x 16B per step). Per-wave async issue count = BYTES/4096.
template <int BYTES>
__device__ __forceinline__ void async_copy_block(const bf16_t* __restrict__ gsrc,
                                                 const bf16_t* lds_dst) {
  const char* g = (const char*)gsrc;
  const unsigned l = (unsigned)(uintptr_t)lds_dst;  // low 32 bits = LDS offset
  const unsigned t16 = (unsigned)threadIdx.x * 16u;
#pragma unroll
  for (int o = 0; o < BYTES; o += 4096) {
    asm volatile("global_load_async_to_lds_b128 %0, %1, off"
                 :
                 : "v"(l + t16 + (unsigned)o), "v"(g + t16 + o)
                 : "memory");
  }
}
#define S_WAIT_ASYNCCNT(n) asm volatile("s_wait_asynccnt " #n ::: "memory")

// ---------------------------------------------------------------------------
// Kernel 1: C = A[M,512] @ W[512,512]^T, output bf16.
//   transposeC==0: C row-major [M,512] (ldC=512)
//   transposeC==1: C transposed  [512,M] (ldC=M)  -- used for V
// Block: 256 thr (8 waves), one 16-row strip of M; wave w owns cols w*64..+63.
// ---------------------------------------------------------------------------
__global__ __launch_bounds__(256)
void proj_kernel(const float* __restrict__ A, const float* __restrict__ W,
                 bf16_t* __restrict__ C, int transposeC, int ldC) {
  const int l0 = blockIdx.x * 16;
  const int w = threadIdx.x >> 5;
  const int lane = threadIdx.x & 31;
  const int hi = lane >> 4, n = lane & 15;

  v8f acc[4] = {};
  for (int k0 = 0; k0 < HIDDEN; k0 += 32) {
    v16bf a = load_a_frag_f32(A + (size_t)l0 * HIDDEN + k0, HIDDEN);
#pragma unroll
    for (int t = 0; t < 4; ++t) {
      v16bf b = load_b_frag_f32(W + (size_t)(w * 64 + t * 16) * HIDDEN + k0, HIDDEN);
      acc[t] = WMMA_BF16(a, b, acc[t]);
    }
  }
#pragma unroll
  for (int t = 0; t < 4; ++t) {
    const int n0 = w * 64 + t * 16;
#pragma unroll
    for (int i = 0; i < 8; ++i) {
      const int m = i + 8 * hi;
      bf16_t v = (bf16_t)acc[t][i];
      if (!transposeC) C[(size_t)(l0 + m) * ldC + n0 + n] = v;
      else             C[(size_t)(n0 + n) * ldC + l0 + m] = v;
    }
  }
}

// ---------------------------------------------------------------------------
// Kernel 2: softmax stats. Block = one 16-row L tile; wave h = head h.
// K strips (16x512 bf16 = 16KB, contiguous) double-buffered in LDS via
// global_load_async_to_lds_b128; online max/sum over all 4096 columns.
// ---------------------------------------------------------------------------
__global__ __launch_bounds__(256)
void attn_stats_kernel(const bf16_t* __restrict__ Qb, const bf16_t* __restrict__ Kb,
                       const float* __restrict__ bias,
                       float* __restrict__ m_ws, float* __restrict__ s_ws) {
  __shared__ bf16_t kstage[2][16 * HIDDEN];  // 2 x 16KB

  const int l0 = blockIdx.x * 16;
  const int h = threadIdx.x >> 5;
  const int lane = threadIdx.x & 31;
  const int hi = lane >> 4, n = lane & 15;

  v16bf a0 = load_a_frag_bf16(Qb + (size_t)l0 * HIDDEN + h * D_HEAD,      HIDDEN);
  v16bf a1 = load_a_frag_bf16(Qb + (size_t)l0 * HIDDEN + h * D_HEAD + 32, HIDDEN);

  float rm[8], rs[8];
#pragma unroll
  for (int i = 0; i < 8; ++i) { rm[i] = -INFINITY; rs[i] = 0.f; }

  async_copy_block<16 * HIDDEN * 2>(Kb, kstage[0]);  // prologue: strip 0

  for (int it = 0; it < NR / 16; ++it) {
    const int buf = it & 1;
    const int r0 = it * 16;
    const bool more = (it + 1 < NR / 16);
    if (more)
      async_copy_block<16 * HIDDEN * 2>(Kb + (size_t)(it + 1) * 16 * HIDDEN,
                                        kstage[buf ^ 1]);
    if (more) S_WAIT_ASYNCCNT(4); else S_WAIT_ASYNCCNT(0);
    __syncthreads();  // current strip resident for all waves

    v16bf b0 = load_b_frag_bf16(kstage[buf] + h * D_HEAD,      HIDDEN);
    v16bf b1 = load_b_frag_bf16(kstage[buf] + h * D_HEAD + 32, HIDDEN);
    v8f acc = {};
    acc = WMMA_BF16(a0, b0, acc);
    acc = WMMA_BF16(a1, b1, acc);

    const float* bp = bias + (size_t)(l0 + 8 * hi) * NR + r0 + n;
    __builtin_prefetch(bp + 16, 0, 1);  // next bias tile -> global_prefetch_b8
#pragma unroll
    for (int i = 0; i < 8; ++i) {
      float x = acc[i] * ATT_SCALE + bp[(size_t)i * NR];
      float t = x;  // max over the 16 lanes of this half-wave (one matrix row)
      for (int d = 8; d >= 1; d >>= 1) t = fmaxf(t, __shfl_xor(t, d, 32));
      float mn = fmaxf(rm[i], t);
      float e = __expf(x - mn);
      float es = e;
      for (int d = 8; d >= 1; d >>= 1) es += __shfl_xor(es, d, 32);
      rs[i] = rs[i] * __expf(rm[i] - mn) + es;
      rm[i] = mn;
    }
    __syncthreads();  // all reads of kstage[buf] done before it is re-filled
  }

  if ((lane & 15) == 0) {
#pragma unroll
    for (int i = 0; i < 8; ++i) {
      const int row = l0 + i + 8 * hi;
      m_ws[(size_t)h * NL + row] = rm[i];
      s_ws[(size_t)h * NL + row] = rs[i];
    }
  }
}

// ---------------------------------------------------------------------------
// Kernel 3: P = exp(s*QK^T + bias - m)/s ; ctx += P @ V ; attn_mean += P/8.
// Block = one 16-row L tile; wave h = head h. K strips (32x512 bf16 = 32KB)
// double-buffered via async copies; heads summed in LDS with ds_add_f32.
// ---------------------------------------------------------------------------
__global__ __launch_bounds__(256)
void attn_ctx_kernel(const bf16_t* __restrict__ Qb, const bf16_t* __restrict__ Kb,
                     const bf16_t* __restrict__ Vt, const float* __restrict__ bias,
                     const float* __restrict__ m_ws, const float* __restrict__ s_ws,
                     bf16_t* __restrict__ ctxb, float* __restrict__ attn_mean) {
  __shared__ bf16_t kstage[2][32 * HIDDEN];   // 2 x 32KB
  __shared__ float  pmean[16 * 32];           // head-summed 16x32 P chunk
  __shared__ bf16_t pstage[HEADS][16 * 32];   // per-wave P tile for re-frag

  const int l0 = blockIdx.x * 16;
  const int h = threadIdx.x >> 5;
  const int lane = threadIdx.x & 31;
  const int hi = lane >> 4, n = lane & 15;

  for (int e = threadIdx.x; e < 512; e += 256) pmean[e] = 0.f;

  v16bf a0 = load_a_frag_bf16(Qb + (size_t)l0 * HIDDEN + h * D_HEAD,      HIDDEN);
  v16bf a1 = load_a_frag_bf16(Qb + (size_t)l0 * HIDDEN + h * D_HEAD + 32, HIDDEN);

  float mrow[8], isrow[8];
#pragma unroll
  for (int i = 0; i < 8; ++i) {
    const int row = l0 + i + 8 * hi;
    mrow[i] = m_ws[(size_t)h * NL + row];
    isrow[i] = 1.f / s_ws[(size_t)h * NL + row];
  }

  v8f c[4] = {};  // ctx accumulators: 16 rows x 64 cols (4 d-tiles)

  async_copy_block<32 * HIDDEN * 2>(Kb, kstage[0]);  // prologue: strip 0

  for (int it = 0; it < NR / 32; ++it) {
    const int buf = it & 1;
    const int r0 = it * 32;
    const bool more = (it + 1 < NR / 32);
    if (more)
      async_copy_block<32 * HIDDEN * 2>(Kb + (size_t)(it + 1) * 32 * HIDDEN,
                                        kstage[buf ^ 1]);
    if (more) S_WAIT_ASYNCCNT(8); else S_WAIT_ASYNCCNT(0);
    __syncthreads();  // strip resident + pmean zeroed/reset

#pragma unroll
    for (int sub = 0; sub < 2; ++sub) {
      const int r1 = r0 + sub * 16;
      const bf16_t* ks = kstage[buf] + (size_t)sub * 16 * HIDDEN;
      v16bf b0 = load_b_frag_bf16(ks + h * D_HEAD,      HIDDEN);
      v16bf b1 = load_b_frag_bf16(ks + h * D_HEAD + 32, HIDDEN);
      v8f acc = {};
      acc = WMMA_BF16(a0, b0, acc);
      acc = WMMA_BF16(a1, b1, acc);

      const float* bp = bias + (size_t)(l0 + 8 * hi) * NR + r1 + n;
      __builtin_prefetch(bp + 32, 0, 1);
#pragma unroll
      for (int i = 0; i < 8; ++i) {
        float x = acc[i] * ATT_SCALE + bp[(size_t)i * NR];
        float p = __expf(x - mrow[i]) * isrow[i];
        const int idx = (i + 8 * hi) * 32 + sub * 16 + n;
        atomicAdd(&pmean[idx], p);      // ds_add_f32, cross-head sum
        pstage[h][idx] = (bf16_t)p;     // stage for A-fragment gather
      }
    }
    // P (16x32 over r) as WMMA A operand; V^T gives contiguous B fragments.
    v16bf pa = load_a_frag_bf16(&pstage[h][0], 32);
#pragma unroll
    for (int t = 0; t < 4; ++t) {
      v16bf vb = load_b_frag_bf16(Vt + (size_t)(h * D_HEAD + t * 16) * NR + r0, NR);
      c[t] = WMMA_BF16(pa, vb, c[t]);
    }
    __syncthreads();  // all heads' ds_add_f32 complete
    for (int e = threadIdx.x; e < 512; e += 256) {
      attn_mean[(size_t)(l0 + (e >> 5)) * NR + r0 + (e & 31)] = pmean[e] * 0.125f;
      pmean[e] = 0.f;
    }
    __syncthreads();  // reset visible; kstage[buf] reads done before re-fill
  }

#pragma unroll
  for (int t = 0; t < 4; ++t)
#pragma unroll
    for (int i = 0; i < 8; ++i)
      ctxb[(size_t)(l0 + i + 8 * hi) * HIDDEN + h * D_HEAD + t * 16 + n] = (bf16_t)c[t][i];
}

// ---------------------------------------------------------------------------
// Kernel 4: x = ctx @ Wo^T + bo + q   (pre-LayerNorm residual), f32 out.
// ---------------------------------------------------------------------------
__global__ __launch_bounds__(256)
void outproj_kernel(const bf16_t* __restrict__ ctxb, const float* __restrict__ Wo,
                    const float* __restrict__ bo, const float* __restrict__ qres,
                    float* __restrict__ x_ws) {
  const int l0 = blockIdx.x * 16;
  const int w = threadIdx.x >> 5;
  const int lane = threadIdx.x & 31;
  const int hi = lane >> 4, n = lane & 15;

  v8f acc[4] = {};
  for (int k0 = 0; k0 < HIDDEN; k0 += 32) {
    v16bf a = load_a_frag_bf16(ctxb + (size_t)l0 * HIDDEN + k0, HIDDEN);
#pragma unroll
    for (int t = 0; t < 4; ++t) {
      v16bf b = load_b_frag_f32(Wo + (size_t)(w * 64 + t * 16) * HIDDEN + k0, HIDDEN);
      acc[t] = WMMA_BF16(a, b, acc[t]);
    }
  }
#pragma unroll
  for (int t = 0; t < 4; ++t) {
    const int col = w * 64 + t * 16 + n;
#pragma unroll
    for (int i = 0; i < 8; ++i) {
      const int row = l0 + i + 8 * hi;
      x_ws[(size_t)row * HIDDEN + col] =
          acc[t][i] + bo[col] + qres[(size_t)row * HIDDEN + col];
    }
  }
}

// ---------------------------------------------------------------------------
// Kernel 5: row LayerNorm. One block (8 waves) per row of 512.
// ---------------------------------------------------------------------------
__global__ __launch_bounds__(256)
void ln_kernel(const float* __restrict__ x, const float* __restrict__ gamma,
               const float* __restrict__ beta, float* __restrict__ out) {
  const int row = blockIdx.x;
  const int t = threadIdx.x;
  const float* xr = x + (size_t)row * HIDDEN;
  float v0 = xr[t], v1 = xr[t + 256];
  float s = v0 + v1, q2 = v0 * v0 + v1 * v1;
  for (int d = 16; d >= 1; d >>= 1) { s += __shfl_xor(s, d, 32); q2 += __shfl_xor(q2, d, 32); }
  __shared__ float ssum[8], ssq[8], mv[2];
  if ((t & 31) == 0) { ssum[t >> 5] = s; ssq[t >> 5] = q2; }
  __syncthreads();
  if (t == 0) {
    float S = 0.f, Q = 0.f;
#pragma unroll
    for (int i = 0; i < 8; ++i) { S += ssum[i]; Q += ssq[i]; }
    float mu = S * (1.f / HIDDEN);
    float var = Q * (1.f / HIDDEN) - mu * mu;
    mv[0] = mu; mv[1] = rsqrtf(var + LN_EPS);
  }
  __syncthreads();
  const float mu = mv[0], rs = mv[1];
  out[(size_t)row * HIDDEN + t]       = (v0 - mu) * rs * gamma[t]       + beta[t];
  out[(size_t)row * HIDDEN + t + 256] = (v1 - mu) * rs * gamma[t + 256] + beta[t + 256];
}

// ---------------------------------------------------------------------------
extern "C" void kernel_launch(void* const* d_in, const int* in_sizes, int n_in,
                              void* d_out, int out_size, void* d_ws, size_t ws_size,
                              hipStream_t stream) {
  const float* q    = (const float*)d_in[0];
  const float* kv   = (const float*)d_in[1];
  const float* bias = (const float*)d_in[2];
  const float* Wq   = (const float*)d_in[3];
  const float* Wk   = (const float*)d_in[4];
  const float* Wv   = (const float*)d_in[5];
  const float* Wo   = (const float*)d_in[6];
  const float* bo   = (const float*)d_in[7];
  const float* gamma = (const float*)d_in[8];
  const float* beta  = (const float*)d_in[9];

  float* outp      = (float*)d_out;               // [NL, 512]
  float* attn_mean = outp + (size_t)NL * HIDDEN;  // [NL, NR]

  size_t off = 0;
  auto carve = [&](size_t bytes) {
    void* p = (char*)d_ws + off;
    off += (bytes + 255) & ~(size_t)255;
    return p;
  };
  bf16_t* Qb   = (bf16_t*)carve((size_t)NL * HIDDEN * 2);   // Q  [2048,512] bf16
  bf16_t* Kb   = (bf16_t*)carve((size_t)NR * HIDDEN * 2);   // K  [4096,512] bf16
  bf16_t* Vt   = (bf16_t*)carve((size_t)HIDDEN * NR * 2);   // V^T[512,4096] bf16
  float*  m_ws = (float*)carve((size_t)HEADS * NL * 4);     // softmax row max
  float*  s_ws = (float*)carve((size_t)HEADS * NL * 4);     // softmax row sum
  bf16_t* ctxb = (bf16_t*)carve((size_t)NL * HIDDEN * 2);   // ctx [2048,512] bf16
  float*  x_ws = (float*)carve((size_t)NL * HIDDEN * 4);    // pre-LN residual
  if (ws_size < off) return;  // workspace too small; ~17 MB required

  dim3 blk(256);
  proj_kernel<<<NL / 16, blk, 0, stream>>>(q,  Wq, Qb, 0, HIDDEN);
  proj_kernel<<<NR / 16, blk, 0, stream>>>(kv, Wk, Kb, 0, HIDDEN);
  proj_kernel<<<NR / 16, blk, 0, stream>>>(kv, Wv, Vt, 1, NR);
  attn_stats_kernel<<<NL / 16, blk, 0, stream>>>(Qb, Kb, bias, m_ws, s_ws);
  attn_ctx_kernel<<<NL / 16, blk, 0, stream>>>(Qb, Kb, Vt, bias, m_ws, s_ws, ctxb, attn_mean);
  outproj_kernel<<<NL / 16, blk, 0, stream>>>(ctxb, Wo, bo, q, x_ws);
  ln_kernel<<<NL, blk, 0, stream>>>(x_ws, gamma, beta, outp);
}
